// Transformer_72816875537065
// MI455X (gfx1250) — matrix operand order repaired and checked by reading, not compiled, not run on hardware
//
#include <hip/hip_runtime.h>
#include <cstdint>

typedef __attribute__((ext_vector_type(16))) _Float16 v16h;
typedef __attribute__((ext_vector_type(8)))  _Float16 v8h;
typedef __attribute__((ext_vector_type(8)))  float    v8f;

#define DEV __device__ __forceinline__
#define MTOT 32768   // B*N = 64*512

// ---------------------------------------------------------------- threefry
DEV void threefry2x32(uint32_t k0, uint32_t k1, uint32_t x0, uint32_t x1,
                      uint32_t& o0, uint32_t& o1) {
  uint32_t ks2 = k0 ^ k1 ^ 0x1BD11BDAu;
  x0 += k0; x1 += k1;
#define TF_R(r) { x0 += x1; x1 = (x1 << r) | (x1 >> (32 - r)); x1 ^= x0; }
  TF_R(13) TF_R(15) TF_R(26) TF_R(6)   x0 += k1;  x1 += ks2 + 1u;
  TF_R(17) TF_R(29) TF_R(16) TF_R(24)  x0 += ks2; x1 += k0 + 2u;
  TF_R(13) TF_R(15) TF_R(26) TF_R(6)   x0 += k0;  x1 += k1 + 3u;
  TF_R(17) TF_R(29) TF_R(16) TF_R(24)  x0 += k1;  x1 += ks2 + 4u;
  TF_R(13) TF_R(15) TF_R(26) TF_R(6)   x0 += ks2; x1 += k0 + 5u;
#undef TF_R
  o0 = x0; o1 = x1;
}

// step_keys = jax.random.split(key(42), 512); key(42)=(0,42); split-halves layout.
__global__ void prep_keys_kernel(uint32_t* __restrict__ keys) {
  int i = threadIdx.x;                       // 0..511, one block of 512
  #pragma unroll
  for (int s = 0; s < 2; ++s) {
    int j = 2 * i + s;
    uint32_t o0, o1, w;
    if (j < 512) { threefry2x32(0u, 42u, (uint32_t)j, (uint32_t)(j + 512), o0, o1); w = o0; }
    else         { threefry2x32(0u, 42u, (uint32_t)(j - 512), (uint32_t)j, o0, o1); w = o1; }
    keys[j] = w;
  }
}

// ---------------------------------------------------------------- weight prep
// in: f32 [K,Nc] row-major  ->  out: f16 [Nc,K] (W^T, K contiguous)
__global__ void transpose_f32_to_f16(const float* __restrict__ in,
                                     _Float16* __restrict__ out, int K, int Nc) {
  int idx = blockIdx.x * 256 + threadIdx.x;
  if (idx >= K * Nc) return;
  int n = idx / K, k = idx - n * K;
  out[idx] = (_Float16)in[k * Nc + n];
}

// ---------------------------------------------------------------- embedding
__global__ __launch_bounds__(256) void embed_kernel(
    const float* __restrict__ problems, const float* __restrict__ W,
    const float* __restrict__ bias, float* __restrict__ xf, _Float16* __restrict__ xh) {
  int t = blockIdx.x * 256 + threadIdx.x;       // over MTOT*128
  int r = t >> 7, d = t & 127;
  float v = problems[r * 2 + 0] * W[d] + problems[r * 2 + 1] * W[128 + d] + bias[d];
  xf[t] = v; xh[t] = (_Float16)v;
}

// ---------------------------------------------------------------- WMMA frags
// A-fragment (16x32 f16): lane<16 -> row lane, K [0..15];
// lane>=16 -> row lane-16, K [16..31].
DEV v16h load_afrag(const _Float16* __restrict__ base, int stride, int k, int lane) {
  int row = lane & 15, g = lane >> 4;
  const _Float16* p = base + (size_t)row * stride + k + 16 * g;
  v8h lo = *(const v8h*)p;
  v8h hi = *(const v8h*)(p + 8);
  return __builtin_shufflevector(lo, hi, 0,1,2,3,4,5,6,7,8,9,10,11,12,13,14,15);
}
// B-fragment (32x16 f16) from W^T [N,K]: per lane col=lane&15, 16 contiguous K.
DEV v16h load_bfrag(const _Float16* __restrict__ base, int stride, int k, int lane) {
  int col = lane & 15, g = lane >> 4;
  return *(const v16h*)(base + (size_t)col * stride + k + 16 * g);
}

// ---------------------------------------------------------------- GEMM (WMMA)
// C[M,Nc] = A(f16 [M,K]) x Wt(f16 [Nc,K])^T (+bias)(+relu) -> C16 and/or C32
// block = 256 thr (8 waves) -> 128x64 tile; wave = 32x32 (2x2 WMMA tiles)
__global__ __launch_bounds__(256) void gemm_wmma(
    const _Float16* __restrict__ A, const _Float16* __restrict__ Wt,
    const float* __restrict__ bias, _Float16* __restrict__ C16,
    float* __restrict__ C32, int M, int Nc, int K, int relu) {
  int lane = threadIdx.x & 31, wave = threadIdx.x >> 5;
  int m0 = blockIdx.y * 128 + (wave >> 1) * 32;
  int n0 = blockIdx.x * 64  + (wave & 1) * 32;
  v8f acc00 = {}, acc01 = {}, acc10 = {}, acc11 = {};
  for (int k = 0; k < K; k += 32) {
    v16h a0 = load_afrag(A  + (size_t)m0 * K,        K, k, lane);
    v16h a1 = load_afrag(A  + (size_t)(m0 + 16) * K, K, k, lane);
    v16h b0 = load_bfrag(Wt + (size_t)n0 * K,        K, k, lane);
    v16h b1 = load_bfrag(Wt + (size_t)(n0 + 16) * K, K, k, lane);
    acc00 = __builtin_amdgcn_wmma_f32_16x16x32_f16(false, a0, false, b0, (short)0, acc00, false, false);
    acc01 = __builtin_amdgcn_wmma_f32_16x16x32_f16(false, a0, false, b1, (short)0, acc01, false, false);
    acc10 = __builtin_amdgcn_wmma_f32_16x16x32_f16(false, a1, false, b0, (short)0, acc10, false, false);
    acc11 = __builtin_amdgcn_wmma_f32_16x16x32_f16(false, a1, false, b1, (short)0, acc11, false, false);
  }
  #pragma unroll
  for (int tm = 0; tm < 2; ++tm) {
    #pragma unroll
    for (int tn = 0; tn < 2; ++tn) {
      v8f acc = tm == 0 ? (tn == 0 ? acc00 : acc01) : (tn == 0 ? acc10 : acc11);
      int cn = n0 + tn * 16 + (lane & 15);
      float bv = bias ? bias[cn] : 0.f;
      #pragma unroll
      for (int j = 0; j < 8; ++j) {
        float v = acc[j] + bv;
        if (relu) v = fmaxf(v, 0.f);
        size_t off = (size_t)(m0 + tm * 16 + ((lane >> 4) * 8) + j) * Nc + cn;
        if (C32) C32[off] = v;
        if (C16) C16[off] = (_Float16)v;
      }
    }
  }
}

// ---------------------------------------------------------------- encoder attention
// One workgroup per (b,h). Flash-style over N=512, dk=dv=16. WMMA for S and PV.
__global__ __launch_bounds__(256) void attn_kernel(const _Float16* __restrict__ qkv,
                                                   _Float16* __restrict__ att) {
  __shared__ _Float16 Qs[512][16];
  __shared__ _Float16 Ks[512][16];
  __shared__ _Float16 Vts[16][512];
  __shared__ _Float16 Pst[8][16][32];
  int b = blockIdx.x >> 3, h = blockIdx.x & 7;
  int t = threadIdx.x, lane = t & 31, wave = t >> 5;
  for (int idx = t; idx < 8192; idx += 256) {
    int n = idx >> 4, d = idx & 15;
    size_t base = (size_t)(b * 512 + n) * 384 + h * 16 + d;
    Qs[n][d]  = (_Float16)((float)qkv[base] * 0.25f);   // 1/sqrt(dk)
    Ks[n][d]  = qkv[base + 128];
    Vts[d][n] = qkv[base + 256];
  }
  __syncthreads();
  int lr = lane & 15, g = lane >> 4;
  v8h  z8  = {};
  v16h z16 = {};
  v8f  zacc = {};
  for (int qb = wave; qb < 32; qb += 8) {
    v16h qrow = *(const v16h*)&Qs[qb * 16 + lr][0];     // dk=16: lanes 16..31 (k 16..31) zero
    v16h afrag = g ? z16 : qrow;
    v8f o = {};
    float m[8], l[8];
    #pragma unroll
    for (int j = 0; j < 8; ++j) { m[j] = -1e30f; l[j] = 0.f; }
    for (int ko = 0; ko < 512; ko += 32) {
      v16h b0 = *(const v16h*)&Ks[ko + lr][0];
      v16h b1 = *(const v16h*)&Ks[ko + 16 + lr][0];
      b0 = (lane & 16) ? z16 : b0;                       // dk rows 16..31 are zero
      b1 = (lane & 16) ? z16 : b1;
      v8f s0 = __builtin_amdgcn_wmma_f32_16x16x32_f16(false, afrag, false, b0, (short)0, zacc, false, false);
      v8f s1 = __builtin_amdgcn_wmma_f32_16x16x32_f16(false, afrag, false, b1, (short)0, zacc, false, false);
      #pragma unroll
      for (int j = 0; j < 8; ++j) {
        float v0 = s0[j], v1 = s1[j];
        float rm = fmaxf(v0, v1);
        rm = fmaxf(rm, __shfl_xor(rm, 1)); rm = fmaxf(rm, __shfl_xor(rm, 2));
        rm = fmaxf(rm, __shfl_xor(rm, 4)); rm = fmaxf(rm, __shfl_xor(rm, 8));
        float nm = fmaxf(m[j], rm);
        float sc = __expf(m[j] - nm);
        float p0 = __expf(v0 - nm), p1 = __expf(v1 - nm);
        float rs = p0 + p1;
        rs += __shfl_xor(rs, 1); rs += __shfl_xor(rs, 2);
        rs += __shfl_xor(rs, 4); rs += __shfl_xor(rs, 8);
        l[j] = l[j] * sc + rs;
        m[j] = nm;
        o[j] *= sc;
        int r = j + 8 * g;
        Pst[wave][r][lr]      = (_Float16)p0;            // C-layout -> A-layout via LDS
        Pst[wave][r][16 + lr] = (_Float16)p1;
      }
      asm volatile("s_wait_dscnt 0x0" ::: "memory");     // wave-private LDS transpose
      v8h plo = *(const v8h*)&Pst[wave][lr][16 * g];
      v8h phi = *(const v8h*)&Pst[wave][lr][16 * g + 8];
      v16h pfrag = __builtin_shufflevector(plo, phi, 0,1,2,3,4,5,6,7,8,9,10,11,12,13,14,15);
      v16h vfrag = *(const v16h*)&Vts[lr][ko + 16 * g];
      o = __builtin_amdgcn_wmma_f32_16x16x32_f16(false, pfrag, false, vfrag, (short)0, o, false, false);
    }
    #pragma unroll
    for (int j = 0; j < 8; ++j) {
      int q = qb * 16 + j + 8 * g;
      att[(size_t)(b * 512 + q) * 128 + h * 16 + lr] = (_Float16)(o[j] / l[j]);
    }
  }
}

// ---------------------------------------------------------------- LayerNorm(x + delta)
__global__ __launch_bounds__(256) void ln_kernel(
    const float* __restrict__ resid, const float* __restrict__ delta,
    const float* __restrict__ g, const float* __restrict__ bb,
    float* __restrict__ xout, _Float16* __restrict__ xh) {
  int row = blockIdx.x * 8 + (threadIdx.x >> 5);
  int lane = threadIdx.x & 31;
  const float* pr = resid + (size_t)row * 128;
  const float* pd = delta + (size_t)row * 128;
  float v[4]; float s = 0.f;
  #pragma unroll
  for (int j = 0; j < 4; ++j) { int c = lane * 4 + j; v[j] = pr[c] + pd[c]; s += v[j]; }
  #pragma unroll
  for (int off = 16; off >= 1; off >>= 1) s += __shfl_xor(s, off);
  float mu = s * (1.f / 128.f);
  float var = 0.f;
  #pragma unroll
  for (int j = 0; j < 4; ++j) { float d = v[j] - mu; var += d * d; }
  #pragma unroll
  for (int off = 16; off >= 1; off >>= 1) var += __shfl_xor(var, off);
  float inv = rsqrtf(var * (1.f / 128.f) + 1e-6f);
  #pragma unroll
  for (int j = 0; j < 4; ++j) {
    int c = lane * 4 + j;
    float o = (v[j] - mu) * inv * g[c] + bb[c];
    xout[(size_t)row * 128 + c] = o;
    xh[(size_t)row * 128 + c] = (_Float16)o;
  }
}

// ---------------------------------------------------------------- graph context
__global__ void mean_kernel(const float* __restrict__ x, float* __restrict__ meanx) {
  int b = blockIdx.x, d = threadIdx.x;       // 64 blocks x 128 thr
  float s = 0.f;
  for (int n = 0; n < 512; ++n) s += x[(size_t)(b * 512 + n) * 128 + d];
  meanx[b * 128 + d] = s * (1.f / 512.f);
}
__global__ void gemv128_kernel(const float* __restrict__ in, const float* __restrict__ W,
                               float* __restrict__ out) {
  __shared__ float sh[128];
  sh[threadIdx.x] = in[blockIdx.x * 128 + threadIdx.x];
  __syncthreads();
  float s = 0.f;
  for (int d = 0; d < 128; ++d) s += sh[d] * W[d * 128 + threadIdx.x];
  out[blockIdx.x * 128 + threadIdx.x] = s;
}

// ---------------------------------------------------------------- block reductions (512 thr)
DEV float blockReduceMax(float v, float* red, int t) {
  #pragma unroll
  for (int off = 16; off >= 1; off >>= 1) v = fmaxf(v, __shfl_xor(v, off));
  __syncthreads();
  if ((t & 31) == 0) red[t >> 5] = v;
  __syncthreads();
  float r = red[0];
  #pragma unroll
  for (int w = 1; w < 16; ++w) r = fmaxf(r, red[w]);
  __syncthreads();
  return r;
}
DEV float blockReduceSum(float v, float* red, int t) {
  #pragma unroll
  for (int off = 16; off >= 1; off >>= 1) v += __shfl_xor(v, off);
  __syncthreads();
  if ((t & 31) == 0) red[t >> 5] = v;
  __syncthreads();
  float r = 0.f;
  #pragma unroll
  for (int w = 0; w < 16; ++w) r += red[w];
  __syncthreads();
  return r;
}
DEV int blockArgmax(float v, int i, float* redv, int* redi, int t) {
  #pragma unroll
  for (int off = 16; off >= 1; off >>= 1) {
    float v2 = __shfl_xor(v, off); int i2 = __shfl_xor(i, off);
    if (v2 > v || (v2 == v && i2 < i)) { v = v2; i = i2; }
  }
  __syncthreads();
  if ((t & 31) == 0) { redv[t >> 5] = v; redi[t >> 5] = i; }
  __syncthreads();
  float bv = redv[0]; int bi = redi[0];
  #pragma unroll
  for (int w = 1; w < 16; ++w) {
    float v2 = redv[w]; int i2 = redi[w];
    if (v2 > bv || (v2 == bv && i2 < bi)) { bv = v2; bi = i2; }
  }
  __syncthreads();
  return bi;
}

// ---------------------------------------------------------------- decoder (persistent, 1 WG per batch)
__global__ __launch_bounds__(512) void decoder_kernel(
    const float* __restrict__ gc, const float* __restrict__ x0,
    const _Float16* __restrict__ Kd, const _Float16* __restrict__ Vd,
    const float* __restrict__ kn, const float* __restrict__ dWq,
    const float* __restrict__ dWo, const float* __restrict__ Wvf,
    const float* __restrict__ Wvl, const uint32_t* __restrict__ keys,
    float* __restrict__ probs_out, int* __restrict__ act_out) {
  __shared__ float cq[384], qv[128], outv[128], qdec[128];
  __shared__ float sArr[8][512];
  __shared__ float logitArr[512];
  __shared__ float lastv[128], firstv[128];
  __shared__ float redv[16];
  __shared__ int   redi[16];
  int b = blockIdx.x, t = threadIdx.x;       // t == node index n
  bool taken = false;
  if (t < 128) { lastv[t] = Wvl[t]; firstv[t] = Wvf[t]; }
  __syncthreads();
  for (int i = 0; i < 512; ++i) {
    if (t < 128) cq[t] = gc[b * 128 + t];
    else if (t < 256) cq[t] = lastv[t - 128];
    else if (t < 384) cq[t] = firstv[t - 256];
    __syncthreads();
    if (t < 128) {                                    // q = cq @ dec_Wq
      float s = 0.f;
      for (int k = 0; k < 384; ++k) s += cq[k] * dWq[k * 128 + t];
      qv[t] = s;
    }
    __syncthreads();
    {                                                 // attention scores (8 heads)
      const _Float16* kr = Kd + (size_t)(b * 512 + t) * 128;
      #pragma unroll
      for (int h = 0; h < 8; ++h) {
        float s = 0.f;
        #pragma unroll
        for (int d = 0; d < 16; ++d) s += qv[h * 16 + d] * (float)kr[h * 16 + d];
        sArr[h][t] = taken ? -1e30f : s * 0.25f;
      }
    }
    __syncthreads();
    for (int h = 0; h < 8; ++h) {                     // per-head softmax
      float v = sArr[h][t];
      float mx = blockReduceMax(v, redv, t);
      float e = __expf(v - mx);
      float sm = blockReduceSum(e, redv, t);
      sArr[h][t] = e / sm;
    }
    __syncthreads();
    if (t < 128) {                                    // P @ V
      int h = t >> 4;
      float acc = 0.f;
      const _Float16* vb = Vd + (size_t)(b * 512) * 128 + t;
      for (int n = 0; n < 512; ++n) acc += sArr[h][n] * (float)vb[(size_t)n * 128];
      outv[t] = acc;
    }
    __syncthreads();
    if (t < 128) {                                    // @ dec_Wo
      float s = 0.f;
      for (int c = 0; c < 128; ++c) s += outv[c] * dWo[c * 128 + t];
      qdec[t] = s;
    }
    __syncthreads();
    float lg;
    {                                                 // pointer logits
      const float* kr = kn + (size_t)(b * 512 + t) * 128;
      float s = 0.f;
      for (int d = 0; d < 128; ++d) s += qdec[d] * kr[d];
      lg = taken ? -1e30f : tanhf(s * 0.08838834764831845f) * 10.f;  // /sqrt(128)
      logitArr[t] = lg;
    }
    // JAX gumbel sampling: threefry(key_i, split-halves counter)
    uint32_t k0 = keys[2 * i], k1 = keys[2 * i + 1];
    int e = b * 512 + t;
    uint32_t o0, o1, bits;
    if (e < 16384) { threefry2x32(k0, k1, (uint32_t)e, (uint32_t)(e + 16384), o0, o1); bits = o0; }
    else           { threefry2x32(k0, k1, (uint32_t)(e - 16384), (uint32_t)e, o0, o1); bits = o1; }
    float f = __uint_as_float((bits >> 9) | 0x3f800000u) - 1.0f;   // [0,1)
    float u = f + 1.1754943508222875e-38f;                          // +tiny (JAX)
    float gum = -logf(-logf(u));
    int a = blockArgmax(lg + gum, t, redv, redi, t);
    float lmx = blockReduceMax(lg, redv, t);
    float ee = expf(lg - lmx);
    float ssum = blockReduceSum(ee, redv, t);
    if (t == 0) {
      float pa = expf(logitArr[a] - lmx) / ssum;
      probs_out[(size_t)i * 64 + b] = pa;
      act_out[(size_t)i * 64 + b] = a;
    }
    if (t == a) taken = true;
    if (t < 128) {                                    // new_last = node_context[0][a]
      float nl = x0[(size_t)a * 128 + t];
      lastv[t] = nl;
      if (i == 0) firstv[t] = nl;
    }
    __syncthreads();
  }
}

// ---------------------------------------------------------------- host driver
extern "C" void kernel_launch(void* const* d_in, const int* in_sizes, int n_in,
                              void* d_out, int out_size, void* d_ws, size_t ws_size,
                              hipStream_t stream) {
  (void)in_sizes; (void)n_in; (void)out_size; (void)ws_size;
  const float* problems = (const float*)d_in[0];
  const float* embed_W  = (const float*)d_in[1];
  const float* embed_b  = (const float*)d_in[2];
  const float* enc_Wq   = (const float*)d_in[3];
  const float* enc_Wk   = (const float*)d_in[4];
  const float* enc_Wv   = (const float*)d_in[5];
  const float* enc_Wo   = (const float*)d_in[6];
  const float* ln1g     = (const float*)d_in[7];
  const float* ln1b     = (const float*)d_in[8];
  const float* ffW1     = (const float*)d_in[9];
  const float* ffb1     = (const float*)d_in[10];
  const float* ffW2     = (const float*)d_in[11];
  const float* ffb2     = (const float*)d_in[12];
  const float* ln2g     = (const float*)d_in[13];
  const float* ln2b     = (const float*)d_in[14];
  const float* graph_W  = (const float*)d_in[15];
  const float* node_W   = (const float*)d_in[16];
  const float* dec_Wq   = (const float*)d_in[17];
  const float* dec_Wk   = (const float*)d_in[18];
  const float* dec_Wv   = (const float*)d_in[19];
  const float* dec_Wo   = (const float*)d_in[20];
  const float* W_vf     = (const float*)d_in[21];
  const float* W_vl     = (const float*)d_in[22];

  char* ws = (char*)d_ws;
  size_t off = 0;
  auto alloc = [&](size_t bytes) -> char* {
    char* p = ws + off; off += (bytes + 255) & ~(size_t)255; return p;
  };
  const int M = MTOT;
  _Float16* wqkvT = (_Float16*)alloc((size_t)3 * 384 * 128 * 2);
  _Float16* woT   = (_Float16*)alloc((size_t)3 * 128 * 128 * 2);
  _Float16* ff1T  = (_Float16*)alloc((size_t)3 * 512 * 128 * 2);
  _Float16* ff2T  = (_Float16*)alloc((size_t)3 * 128 * 512 * 2);
  _Float16* nodeT = (_Float16*)alloc((size_t)128 * 128 * 2);
  _Float16* dkT   = (_Float16*)alloc((size_t)128 * 128 * 2);
  _Float16* dvT   = (_Float16*)alloc((size_t)128 * 128 * 2);
  float*    xf    = (float*)   alloc((size_t)M * 128 * 4);
  _Float16* xh    = (_Float16*)alloc((size_t)M * 128 * 2);
  _Float16* qkv   = (_Float16*)alloc((size_t)M * 384 * 2);
  _Float16* att   = (_Float16*)alloc((size_t)M * 128 * 2);
  float*    tmp   = (float*)   alloc((size_t)M * 128 * 4);
  _Float16* hbuf  = (_Float16*)alloc((size_t)M * 512 * 2);
  float*    knode = (float*)   alloc((size_t)M * 128 * 4);
  _Float16* Kd    = (_Float16*)alloc((size_t)M * 128 * 2);
  _Float16* Vd    = (_Float16*)alloc((size_t)M * 128 * 2);
  float*    meanx = (float*)   alloc((size_t)64 * 128 * 4);
  float*    gcb   = (float*)   alloc((size_t)64 * 128 * 4);
  uint32_t* keys  = (uint32_t*)alloc((size_t)1024 * 4);

  prep_keys_kernel<<<1, 512, 0, stream>>>(keys);
  auto T = [&](const float* in, _Float16* out, int K, int Nc) {
    int n = K * Nc;
    transpose_f32_to_f16<<<(n + 255) / 256, 256, 0, stream>>>(in, out, K, Nc);
  };
  for (int l = 0; l < 3; ++l) {
    T(enc_Wq + l * 16384, wqkvT + l * 49152,          128, 128);
    T(enc_Wk + l * 16384, wqkvT + l * 49152 + 16384,  128, 128);
    T(enc_Wv + l * 16384, wqkvT + l * 49152 + 32768,  128, 128);
    T(enc_Wo + l * 16384, woT   + l * 16384,          128, 128);
    T(ffW1   + l * 65536, ff1T  + l * 65536,          128, 512);
    T(ffW2   + l * 65536, ff2T  + l * 65536,          512, 128);
  }
  T(node_W, nodeT, 128, 128);
  T(dec_Wk, dkT,   128, 128);
  T(dec_Wv, dvT,   128, 128);

  embed_kernel<<<(M * 128) / 256, 256, 0, stream>>>(problems, embed_W, embed_b, xf, xh);

  for (int l = 0; l < 3; ++l) {
    gemm_wmma<<<dim3(6, M / 128), 256, 0, stream>>>(xh, wqkvT + l * 49152, nullptr, qkv, nullptr, M, 384, 128, 0);
    attn_kernel<<<512, 256, 0, stream>>>(qkv, att);
    gemm_wmma<<<dim3(2, M / 128), 256, 0, stream>>>(att, woT + l * 16384, nullptr, nullptr, tmp, M, 128, 128, 0);
    ln_kernel<<<M / 8, 256, 0, stream>>>(xf, tmp, ln1g + l * 128, ln1b + l * 128, xf, xh);
    gemm_wmma<<<dim3(8, M / 128), 256, 0, stream>>>(xh, ff1T + l * 65536, ffb1 + l * 512, hbuf, nullptr, M, 512, 128, 1);
    gemm_wmma<<<dim3(2, M / 128), 256, 0, stream>>>(hbuf, ff2T + l * 65536, ffb2 + l * 128, nullptr, tmp, M, 128, 512, 0);
    ln_kernel<<<M / 8, 256, 0, stream>>>(xf, tmp, ln2g + l * 128, ln2b + l * 128, xf, xh);
  }
  mean_kernel<<<64, 128, 0, stream>>>(xf, meanx);
  gemv128_kernel<<<64, 128, 0, stream>>>(meanx, graph_W, gcb);
  gemm_wmma<<<dim3(2, M / 128), 256, 0, stream>>>(xh, nodeT, nullptr, nullptr, knode, M, 128, 128, 0);
  gemm_wmma<<<dim3(2, M / 128), 256, 0, stream>>>(xh, dkT, nullptr, Kd, nullptr, M, 128, 128, 0);
  gemm_wmma<<<dim3(2, M / 128), 256, 0, stream>>>(xh, dvT, nullptr, Vd, nullptr, M, 128, 128, 0);

  float* probs_out = (float*)d_out;
  int* act_out = (int*)(probs_out + 32768);
  decoder_kernel<<<64, 512, 0, stream>>>(gcb, xf, Kd, Vd, knode, dec_Wq, dec_Wo,
                                         W_vf, W_vl, keys, probs_out, act_out);
}
